// QuerySATDecoder_28046136442975
// MI455X (gfx1250) — compile-verified, matrix operand order checked by reading, not run on hardware
//
#include <hip/hip_runtime.h>

typedef __attribute__((ext_vector_type(8)))  float  v8f;
typedef __attribute__((ext_vector_type(16))) __bf16 v16bf;
typedef __attribute__((ext_vector_type(2)))  float  v2f;
typedef __attribute__((ext_vector_type(2)))  __bf16 v2bf;

union Frag {
  v16bf v;
  uint4 q[2];
  unsigned int u[8];
  unsigned short s[16];
};

// scalar RNE f32->bf16 (cold paths)
__device__ __forceinline__ unsigned short f2bf(float f) {
  unsigned int u = __float_as_uint(f);
  unsigned int r = u + 0x7FFFu + ((u >> 16) & 1u);
  return (unsigned short)(r >> 16);
}

// packed pair conversion -> v_cvt_pk_bf16_f32 (hot paths)
__device__ __forceinline__ unsigned int f2bf2(float lo, float hi) {
  union { v2bf b; unsigned int u; } cv;
  v2f f; f.x = lo; f.y = hi;
  cv.b = __builtin_convertvector(f, v2bf);
  return cv.u;
}

// ---------------------------------------------------------------------------
// Init: zero accumulators, set graph masks, and pre-convert W1/W2 to bf16
// (transposed, N-padded) so k_mlp staging is a pure vector copy.
// ---------------------------------------------------------------------------
__global__ void k_init(unsigned int* __restrict__ zbase, size_t zwords,
                       unsigned int* __restrict__ gmask, int n_graphs,
                       const float* __restrict__ W1, const float* __restrict__ W2,
                       unsigned short* __restrict__ w1bf,   // [128][128]: [n][k]
                       unsigned short* __restrict__ w2bf)   // [16][128]:  [n][k]
{
  const size_t stride = (size_t)gridDim.x * blockDim.x;
  const size_t i0 = (size_t)blockIdx.x * blockDim.x + threadIdx.x;
  for (size_t i = i0; i < zwords; i += stride) zbase[i] = 0u;
  if (i0 < (size_t)n_graphs) gmask[i0] = 0xFFu;
  for (size_t i = i0; i < 128 * 128; i += stride) {
    const int k = (int)(i >> 7), n = (int)(i & 127);
    w1bf[n * 128 + k] = f2bf(W1[i]);                  // transpose: [n][k] = W1[k][n]
  }
  for (size_t i = i0; i < 16 * 128; i += stride) {
    const int n = (int)(i >> 7), k = (int)(i & 127);
    w2bf[i] = (n < 8) ? f2bf(W2[k * 8 + n]) : (unsigned short)0;
  }
}

// ---------------------------------------------------------------------------
// MLP: logits = relu(x@W1 + b1) @ W2 + b2 via bf16 WMMA, f32 accumulate.
// Block = 128 threads = 4 waves; each wave owns a 16-row tile per iteration.
// ---------------------------------------------------------------------------
#define HP 136  // padded LDS row stride (bf16 elems): 272B = 68 dwords, conflict-free
__global__ __launch_bounds__(128)
void k_mlp(const float* __restrict__ x,
           const unsigned short* __restrict__ w1bf,
           const unsigned short* __restrict__ w2bf,
           const float* __restrict__ b1, const float* __restrict__ b2,
           float* __restrict__ logits, int n_vars)
{
  __shared__ __align__(16) unsigned short w1t[128 * HP];
  __shared__ __align__(16) unsigned short w2t[16 * HP];
  __shared__ __align__(16) unsigned short hid[4 * 16 * HP];
  __shared__ float b1s[128];

  const int tid = threadIdx.x;
  // vector-copy pre-converted weights into padded LDS
  {
    const uint4* s1 = (const uint4*)w1bf;             // 2048 x 16B, 16 per row
    for (int idx = tid; idx < 2048; idx += blockDim.x) {
      const int row = idx >> 4, c = idx & 15;
      *(uint4*)&w1t[row * HP + c * 8] = s1[idx];
    }
    const uint4* s2 = (const uint4*)w2bf;             // 256 x 16B
    for (int idx = tid; idx < 256; idx += blockDim.x) {
      const int row = idx >> 4, c = idx & 15;
      *(uint4*)&w2t[row * HP + c * 8] = s2[idx];
    }
    if (tid < 128) b1s[tid] = b1[tid];
  }
  __syncthreads();

  const int lane = tid & 31;
  const int wave = tid >> 5;
  const int r    = lane & 15;   // row (A) or column (B/D) within tile
  const int half = lane >> 4;   // K-half selector per ISA bf16 layout
  unsigned short* hrow = hid + wave * 16 * HP;

  const int n_tiles = (n_vars + 15) >> 4;
  for (int tile = blockIdx.x * 4 + wave; tile < n_tiles; tile += gridDim.x * 4) {
    const int row0 = tile << 4;
    int rrow = row0 + r; if (rrow > n_vars - 1) rrow = n_vars - 1;
    const float* xr = x + (size_t)rrow * 128;

    // ---- A fragments: 16x32 bf16 per K-chunk, straight from global x
    Frag a[4];
    #pragma unroll
    for (int kc = 0; kc < 4; ++kc) {
      const int k0 = kc * 32 + half * 8;
      float4 f0 = *(const float4*)(xr + k0);
      float4 f1 = *(const float4*)(xr + k0 + 4);
      float4 g0 = *(const float4*)(xr + k0 + 16);
      float4 g1 = *(const float4*)(xr + k0 + 20);
      a[kc].u[0] = f2bf2(f0.x, f0.y);
      a[kc].u[1] = f2bf2(f0.z, f0.w);
      a[kc].u[2] = f2bf2(f1.x, f1.y);
      a[kc].u[3] = f2bf2(f1.z, f1.w);
      a[kc].u[4] = f2bf2(g0.x, g0.y);
      a[kc].u[5] = f2bf2(g0.z, g0.w);
      a[kc].u[6] = f2bf2(g1.x, g1.y);
      a[kc].u[7] = f2bf2(g1.z, g1.w);
    }

    // ---- Layer 1: 8 N-tiles; batch all 4 B-fragments then burst 4 WMMAs
    #pragma unroll
    for (int n = 0; n < 8; ++n) {
      Frag b[4];
      #pragma unroll
      for (int kc = 0; kc < 4; ++kc) {
        const unsigned short* bp = w1t + (n * 16 + r) * HP + kc * 32 + half * 8;
        b[kc].q[0] = *(const uint4*)bp;
        b[kc].q[1] = *(const uint4*)(bp + 16);
      }
      v8f acc = {};
      #pragma unroll
      for (int kc = 0; kc < 4; ++kc)
        acc = __builtin_amdgcn_wmma_f32_16x16x32_bf16(
            false, a[kc].v, false, b[kc].v, (short)0, acc, false, false);

      const int colg = n * 16 + r;
      const float bias = b1s[colg];
      #pragma unroll
      for (int v = 0; v < 8; v += 2) {   // D: row = v + half*8, col = lane&15
        const float h0 = fmaxf(acc[v] + bias, 0.f);
        const float h1 = fmaxf(acc[v + 1] + bias, 0.f);
        const unsigned int u = f2bf2(h0, h1);
        hrow[(v + half * 8) * HP + colg]     = (unsigned short)u;
        hrow[(v + 1 + half * 8) * HP + colg] = (unsigned short)(u >> 16);
      }
    }

    // ---- Layer 2: hidden(16x128) @ W2(128x16-padded); batch loads, 4 WMMAs
    Frag a2[4], bw[4];
    #pragma unroll
    for (int kc = 0; kc < 4; ++kc) {
      const unsigned short* ap = hrow + r * HP + kc * 32 + half * 8;
      a2[kc].q[0] = *(const uint4*)ap;
      a2[kc].q[1] = *(const uint4*)(ap + 16);
      const unsigned short* bp = w2t + r * HP + kc * 32 + half * 8;
      bw[kc].q[0] = *(const uint4*)bp;
      bw[kc].q[1] = *(const uint4*)(bp + 16);
    }
    v8f acc2 = {};
    #pragma unroll
    for (int kc = 0; kc < 4; ++kc)
      acc2 = __builtin_amdgcn_wmma_f32_16x16x32_bf16(
          false, a2[kc].v, false, bw[kc].v, (short)0, acc2, false, false);

    if (r < 8) {                         // only first 8 output columns are real
      const float bias2 = b2[r];
      #pragma unroll
      for (int v = 0; v < 8; ++v) {
        const int rr = row0 + v + half * 8;
        if (rr < n_vars) logits[(size_t)rr * 8 + r] = acc2[v] + bias2;
      }
    }
  }
}

// ---------------------------------------------------------------------------
// Edge pass: gather logits, softplus scatter-add into clause_sum,
// hard assignment as a per-clause 8-bit OR mask. Prefetch next gather row.
// ---------------------------------------------------------------------------
__global__ __launch_bounds__(256)
void k_edges(const float* __restrict__ logits, const float* __restrict__ pol,
             const int* __restrict__ vidx, const int* __restrict__ cidx,
             float* __restrict__ csum, unsigned int* __restrict__ cbits,
             int n_edges)
{
  const size_t stride = (size_t)gridDim.x * blockDim.x;
  for (size_t e = (size_t)blockIdx.x * blockDim.x + threadIdx.x;
       e < (size_t)n_edges; e += stride) {
    const size_t en = e + stride;
    if (en < (size_t)n_edges)
      __builtin_prefetch(logits + (size_t)vidx[en] * 8, 0, 0);  // global_prefetch_b8

    const int vi = vidx[e];
    const int ci = cidx[e];
    const float p = pol[e];
    const float4 l0 = *(const float4*)(logits + (size_t)vi * 8);
    const float4 l1 = *(const float4*)(logits + (size_t)vi * 8 + 4);
    const float l[8] = {l0.x, l0.y, l0.z, l0.w, l1.x, l1.y, l1.z, l1.w};
    float* cs = csum + (size_t)ci * 8;
    const bool pp = p > 0.f;
    unsigned int mask = 0;
    #pragma unroll
    for (int k = 0; k < 8; ++k) {
      const float z = l[k] * p;
      const float sp = fmaxf(z, 0.f) + log1pf(__expf(-fabsf(z))); // stable softplus
      atomicAdd(cs + k, sp);
      if ((l[k] > 0.f) == pp) mask |= (1u << k);
    }
    atomicOr(cbits + ci, mask);
  }
}

// ---------------------------------------------------------------------------
// Clause pass: per-clause loss, LDS-staged segment-sum into graphs,
// AND-reduce of hard masks per graph.
// ---------------------------------------------------------------------------
__global__ __launch_bounds__(256)
void k_clauses(const float* __restrict__ csum, const unsigned int* __restrict__ cbits,
               const int* __restrict__ cbatch, float* __restrict__ pgl,
               unsigned int* __restrict__ gmask, int n_clauses, int n_graphs)
{
  __shared__ float sloss[256 * 8];
  __shared__ unsigned int smask[256];
  const int tid = threadIdx.x;
  #pragma unroll
  for (int i = 0; i < 8; ++i) sloss[tid * 8 + i] = 0.f;
  smask[tid] = 0xFFu;
  __syncthreads();

  const size_t stride = (size_t)gridDim.x * blockDim.x;
  for (size_t c = (size_t)blockIdx.x * blockDim.x + tid;
       c < (size_t)n_clauses; c += stride) {
    const int g = cbatch[c];
    const float4 s0 = *(const float4*)(csum + c * 8);
    const float4 s1 = *(const float4*)(csum + c * 8 + 4);
    const float s[8] = {s0.x, s0.y, s0.z, s0.w, s1.x, s1.y, s1.z, s1.w};
    float* sl = sloss + g * 8;
    #pragma unroll
    for (int k = 0; k < 8; ++k) {
      const float cv = __expf(-s[k]);
      const float loss = cv * (-__logf(1.f - cv + 1e-8f));
      atomicAdd(sl + k, loss);               // ds_add_f32
    }
    atomicAnd(&smask[g], cbits[c]);
  }
  __syncthreads();
  for (int i = tid; i < 256 * 8; i += 256) {
    const float v = sloss[i];
    if (v != 0.f) atomicAdd(&pgl[i], v);
  }
  if (tid < n_graphs) {
    const unsigned int m = smask[tid];
    if (m != 0xFFu) atomicAnd(&gmask[tid], m);
  }
}

// ---------------------------------------------------------------------------
// Final: sqrt-shift, per-graph descending sort of 8 vals, cost-weighted sum,
// block reduce; solved flags from graph masks.
// ---------------------------------------------------------------------------
__global__ __launch_bounds__(256)
void k_final(const float* __restrict__ pgl, const unsigned int* __restrict__ gmask,
             float* __restrict__ out, int n_graphs)
{
  __shared__ float red[256];
  const int g = threadIdx.x;
  float dot = 0.f;
  if (g < n_graphs) {
    float a[8];
    const float s0 = sqrtf(1e-6f);
    #pragma unroll
    for (int k = 0; k < 8; ++k) a[k] = sqrtf(pgl[g * 8 + k] + 1e-6f) - s0;
    for (int i = 1; i < 8; ++i) {          // insertion sort, descending
      const float key = a[i];
      int j = i - 1;
      while (j >= 0 && a[j] < key) { a[j + 1] = a[j]; --j; }
      a[j + 1] = key;
    }
    #pragma unroll
    for (int k = 0; k < 8; ++k) dot += a[k] * (float)((k + 1) * (k + 1));
    out[1 + g] = (gmask[g] != 0u) ? 1.0f : 0.0f;
  }
  red[g] = dot;
  __syncthreads();
  for (int off = 128; off > 0; off >>= 1) {
    if (g < off) red[g] += red[g + off];
    __syncthreads();
  }
  if (g == 0) out[0] = red[0] / 204.0f;    // sum_{i=1..8} i^2 = 204
}

// ---------------------------------------------------------------------------
extern "C" void kernel_launch(void* const* d_in, const int* in_sizes, int n_in,
                              void* d_out, int out_size, void* d_ws, size_t ws_size,
                              hipStream_t stream)
{
  const float* x   = (const float*)d_in[0];
  const float* pol = (const float*)d_in[1];
  const int*   vix = (const int*)d_in[2];
  const int*   cix = (const int*)d_in[3];
  const int*   cbt = (const int*)d_in[4];
  const float* W1  = (const float*)d_in[5];
  const float* b1  = (const float*)d_in[6];
  const float* W2  = (const float*)d_in[7];
  const float* b2  = (const float*)d_in[8];

  const int dim       = in_sizes[6];            // 128
  const int n_vars    = in_sizes[0] / dim;      // 500000
  const int n_edges   = in_sizes[1];            // 6000000
  const int n_clauses = in_sizes[4];            // 2000000
  const int n_graphs  = out_size - 1;           // 256

  auto align256 = [](size_t v) { return (v + 255) & ~(size_t)255; };
  char* ws = (char*)d_ws;
  size_t off = 0;
  float* logits = (float*)(ws + off);
  off = align256(off + (size_t)n_vars * 8 * sizeof(float));
  const size_t off_zero = off;
  float* csum = (float*)(ws + off);
  off = align256(off + (size_t)n_clauses * 8 * sizeof(float));
  unsigned int* cbits = (unsigned int*)(ws + off);
  off = align256(off + (size_t)n_clauses * sizeof(unsigned int));
  float* pgl = (float*)(ws + off);
  off = align256(off + (size_t)n_graphs * 8 * sizeof(float));
  const size_t off_zero_end = off;
  unsigned int* gmask = (unsigned int*)(ws + off);
  off = align256(off + (size_t)n_graphs * sizeof(unsigned int));
  unsigned short* w1bf = (unsigned short*)(ws + off);
  off = align256(off + (size_t)128 * 128 * sizeof(unsigned short));
  unsigned short* w2bf = (unsigned short*)(ws + off);
  off = align256(off + (size_t)16 * 128 * sizeof(unsigned short));
  (void)ws_size; (void)n_in;

  const size_t zwords = (off_zero_end - off_zero) / 4;
  hipLaunchKernelGGL(k_init, dim3(2048), dim3(256), 0, stream,
                     (unsigned int*)(ws + off_zero), zwords, gmask, n_graphs,
                     W1, W2, w1bf, w2bf);
  hipLaunchKernelGGL(k_mlp, dim3(2048), dim3(128), 0, stream,
                     x, w1bf, w2bf, b1, b2, logits, n_vars);
  hipLaunchKernelGGL(k_edges, dim3(4096), dim3(256), 0, stream,
                     logits, pol, vix, cix, csum, cbits, n_edges);
  hipLaunchKernelGGL(k_clauses, dim3(2048), dim3(256), 0, stream,
                     csum, cbits, cbt, pgl, gmask, n_clauses, n_graphs);
  hipLaunchKernelGGL(k_final, dim3(1), dim3(256), 0, stream,
                     pgl, gmask, (float*)d_out, n_graphs);
}
